// GCN_59657095741747
// MI455X (gfx1250) — matrix operand order repaired and checked by reading, not compiled, not run on hardware
//
#include <hip/hip_runtime.h>

// ---------------------------------------------------------------------------
// GCN forward on MI455X (gfx1250), bf16 WMMA with fp32 accumulate.
//   per layer:  agg = (adj @ z) * (1/deg)   [8192x8192x256 GEMM, HBM-bound]
//               z   = tanh(agg @ W + z @ B) [two 8192x256x256 GEMMs]
// B tiles staged by the Tensor Data Mover (tensor_load_to_lds) when available;
// A tiles double-buffered through registers (global loads overlap compute).
// ---------------------------------------------------------------------------

#define NN   8192   // nodes
#define DD   256    // embedding dim
#define BKP  40     // LDS row pitch in bf16 (80 B = 64 B data + 16 B pad)

typedef __attribute__((ext_vector_type(16))) __bf16 bf16x16;
typedef __attribute__((ext_vector_type(8)))  __bf16 bf16x8;
typedef __attribute__((ext_vector_type(4)))  __bf16 bf16x4;
typedef __attribute__((ext_vector_type(8)))  float  f32x8;
typedef __attribute__((ext_vector_type(4)))  unsigned int u32x4;
typedef __attribute__((ext_vector_type(8)))  int  i32x8;
typedef __attribute__((ext_vector_type(4)))  int  i32x4;

#if defined(__has_builtin)
#if __has_builtin(__builtin_amdgcn_tensor_load_to_lds) && \
    __has_builtin(__builtin_amdgcn_s_wait_tensorcnt)
#define USE_TDM 1
#endif
#endif

// -------------------- fp32 -> bf16 transpose (tiled, coalesced) ------------
__global__ __launch_bounds__(256) void transpose_f32_to_bf16(
    const float* __restrict__ in, __bf16* __restrict__ out, int R, int C)
{
    __shared__ float tile[32][33];
    const int tx = threadIdx.x & 31;
    const int ty = threadIdx.x >> 5;
    const int r0 = blockIdx.y * 32;
    const int c0 = blockIdx.x * 32;
#pragma unroll
    for (int j = 0; j < 4; ++j)
        tile[ty + j * 8][tx] = in[(long)(r0 + ty + j * 8) * C + c0 + tx];
    __syncthreads();
#pragma unroll
    for (int j = 0; j < 4; ++j)
        out[(long)(c0 + ty + j * 8) * R + r0 + tx] = (__bf16)tile[tx][ty + j * 8];
}

// -------------------- TDM: 2D bf16 tile (32 k x 256 rows) -> padded LDS ----
#ifdef USE_TDM
__device__ __forceinline__ void tdm_load_tile_bf16(
    unsigned lds_off, const __bf16* gsrc, unsigned tdim0, unsigned stride0)
{
    const unsigned long long ga = (unsigned long long)gsrc;
    u32x4 g0;
    g0[0] = 1u;                                   // count=1, user mode, no gather
    g0[1] = lds_off;                              // LDS byte address
    g0[2] = (unsigned)(ga & 0xffffffffu);         // global_addr[31:0]
    g0[3] = (unsigned)((ga >> 32) & 0x01ffffffu)  // global_addr[56:32]
          | (2u << 30);                           // type = 2 ("image")
    i32x8 g1;
    g1[0] = (int)((1u << 16)      // data_size = 2 bytes
                | (1u << 20)      // pad_enable (load only)
                | (3u << 22)      // pad_interval: 16 DWORDs (= 64 B row)
                | (3u << 25));    // pad_amount:   4 DWORDs (= 16 B pad)
    g1[1] = (int)((tdim0 & 0xffffu) << 16);                    // tensor_dim0 lo
    g1[2] = (int)(((tdim0 >> 16) & 0xffffu) | (256u << 16));   // dim0 hi | tensor_dim1=256
    g1[3] = (int)(32u << 16);                                  // tile_dim0 = 32
    g1[4] = (int)256u;                                         // tile_dim1 = 256
    g1[5] = (int)stride0;                                      // tensor_dim0_stride lo
    g1[6] = 0;
    g1[7] = 0;
    const i32x4 gz4 = {0, 0, 0, 0};
    const i32x8 gz8 = {0, 0, 0, 0, 0, 0, 0, 0};
    __builtin_amdgcn_tensor_load_to_lds(g0, g1, gz4, gz4, gz8, 0);
}
#endif

// -------------------- fallback VALU staging helpers ------------------------
__device__ __forceinline__ void stage_b(__bf16* sB, const __bf16* __restrict__ src,
                                        long ld, int k0, int tid)
{
    for (int c = tid; c < 256 * 4; c += 128) {
        const int n  = c >> 2;
        const int cc = (c & 3) << 3;
        bf16x8 v = *(const bf16x8*)(src + (long)n * ld + k0 + cc);
        *(bf16x8*)(sB + n * BKP + cc) = v;
    }
}

// -------------------- register double-buffer for A tiles -------------------
__device__ __forceinline__ void load_a_regs(float4 r[4], const float* __restrict__ src,
                                            long ld, int m0, int k0, int tid)
{
#pragma unroll
    for (int i = 0; i < 4; ++i) {
        const int c  = tid + i * 128;
        const int rr = c >> 3, c4 = (c & 7) << 2;
        r[i] = *(const float4*)(src + (long)(m0 + rr) * ld + k0 + c4);
    }
}

__device__ __forceinline__ void store_a_regs(__bf16* sA, const float4 r[4], int tid)
{
#pragma unroll
    for (int i = 0; i < 4; ++i) {
        const int c  = tid + i * 128;
        const int rr = c >> 3, c4 = (c & 7) << 2;
        bf16x4 h;
        h[0] = (__bf16)r[i].x; h[1] = (__bf16)r[i].y;
        h[2] = (__bf16)r[i].z; h[3] = (__bf16)r[i].w;
        *(bf16x4*)(sA + rr * BKP + c4) = h;
    }
}

// -------------------- WMMA fragment loads (ISA 16-bit layouts) -------------
__device__ __forceinline__ bf16x16 load_a_frag(const __bf16* sA, int mbase, int lane)
{
    const int m    = mbase + (lane & 15);
    const int koff = (lane >> 4) * 8;
    const __bf16* p = sA + m * BKP + koff;
    bf16x8 lo = *(const bf16x8*)p;
    bf16x8 hi = *(const bf16x8*)(p + 16);
    bf16x16 a;
#pragma unroll
    for (int i = 0; i < 8; ++i) { a[i] = lo[i]; a[i + 8] = hi[i]; }
    return a;
}

__device__ __forceinline__ bf16x16 load_b_frag(const __bf16* sB, int n0, int lane)
{
    const int n  = n0 + (lane & 15);
    const int kb = (lane >> 4) * 8;
    const __bf16* p = sB + n * BKP + kb;
    bf16x8 b0 = *(const bf16x8*)p;
    bf16x8 b1 = *(const bf16x8*)(p + 16);
    bf16x16 b;
#pragma unroll
    for (int i = 0; i < 8; ++i) { b[i] = b0[i]; b[i + 8] = b1[i]; }
    return b;
}

// -------------------- kernel A: agg = (adj @ z) * (1/deg) ------------------
__global__ __launch_bounds__(128) void gcn_agg_wmma(
    const float*  __restrict__ adj,     // [NN][NN] fp32
    const __bf16* __restrict__ zT,      // [DD][NN] bf16 (transposed z)
    const float*  __restrict__ degree,  // [NN]
    float*        __restrict__ agg)     // [NN][DD] fp32
{
    __shared__ __bf16 sA[64 * BKP];
    __shared__ __bf16 sB[256 * BKP];
    const int tid  = threadIdx.x;
    const int lane = tid & 31;
    const int wave = tid >> 5;
    const int m0   = blockIdx.x * 64;

    f32x8 acc[16];
#pragma unroll
    for (int t = 0; t < 16; ++t)
#pragma unroll
        for (int v = 0; v < 8; ++v) acc[t][v] = 0.0f;

    float4 aregs[4];
    load_a_regs(aregs, adj, NN, m0, 0, tid);     // prologue: first A tile

#pragma unroll 1
    for (int k0 = 0; k0 < NN; k0 += 32) {
        __syncthreads();                          // LDS free for new tiles
        store_a_regs(sA, aregs, tid);
#ifdef USE_TDM
        if (wave == 0) {                          // one wave drives the DMA
            tdm_load_tile_bf16((unsigned)(unsigned long long)sB,
                               zT + k0, NN, NN);
            __builtin_amdgcn_s_wait_tensorcnt(0);
        }
#else
        stage_b(sB, zT, NN, k0, tid);
#endif
        __syncthreads();                          // tiles published
        if (k0 + 32 < NN)                         // next A tile in flight
            load_a_regs(aregs, adj, NN, m0, k0 + 32, tid);

        const bf16x16 a = load_a_frag(sA, wave * 16, lane);
        bf16x16 bc = load_b_frag(sB, 0, lane);    // pipelined B fragments
#pragma unroll
        for (int nt = 0; nt < 16; ++nt) {
            bf16x16 bn = bc;
            if (nt < 15) bn = load_b_frag(sB, (nt + 1) * 16, lane);
            acc[nt] = __builtin_amdgcn_wmma_f32_16x16x32_bf16(
                false, a, false, bc, (short)0, acc[nt], false, false);
            bc = bn;
        }
    }

    // epilogue: C layout -> lane = N (per half), VGPR v = M row; scale by 1/deg
    const int n  = lane & 15;
    const int mb = (lane >> 4) * 8;
    float s[8];
#pragma unroll
    for (int v = 0; v < 8; ++v) s[v] = 1.0f / degree[m0 + wave * 16 + mb + v];
#pragma unroll
    for (int nt = 0; nt < 16; ++nt)
#pragma unroll
        for (int v = 0; v < 8; ++v) {
            const int row = m0 + wave * 16 + mb + v;
            agg[(long)row * DD + nt * 16 + n] = acc[nt][v] * s[v];
        }
}

// -------------------- kernel B: z = tanh(agg @ W + z @ B) ------------------
__global__ __launch_bounds__(128) void gcn_update_wmma(
    const float*  __restrict__ agg,   // [NN][DD] fp32
    const float*  __restrict__ zin,   // [NN][DD] fp32
    const __bf16* __restrict__ wT,    // [DD][DD] bf16 (transposed W)
    const __bf16* __restrict__ bT,    // [DD][DD] bf16 (transposed B)
    float*        __restrict__ zout)  // [NN][DD] fp32
{
    __shared__ __bf16 sA1[64 * BKP];
    __shared__ __bf16 sA2[64 * BKP];
    __shared__ __bf16 sB1[256 * BKP];
    __shared__ __bf16 sB2[256 * BKP];
    const int tid  = threadIdx.x;
    const int lane = tid & 31;
    const int wave = tid >> 5;
    const int m0   = blockIdx.x * 64;

    f32x8 acc[16];
#pragma unroll
    for (int t = 0; t < 16; ++t)
#pragma unroll
        for (int v = 0; v < 8; ++v) acc[t][v] = 0.0f;

    float4 r1[4], r2[4];
    load_a_regs(r1, agg, DD, m0, 0, tid);
    load_a_regs(r2, zin, DD, m0, 0, tid);

#pragma unroll 1
    for (int k0 = 0; k0 < DD; k0 += 32) {
        __syncthreads();
        store_a_regs(sA1, r1, tid);
        store_a_regs(sA2, r2, tid);
#ifdef USE_TDM
        if (wave == 0) {
            tdm_load_tile_bf16((unsigned)(unsigned long long)sB1, wT + k0, DD, DD);
            __builtin_amdgcn_s_wait_tensorcnt(0);
        } else if (wave == 1) {
            tdm_load_tile_bf16((unsigned)(unsigned long long)sB2, bT + k0, DD, DD);
            __builtin_amdgcn_s_wait_tensorcnt(0);
        }
#else
        stage_b(sB1, wT, DD, k0, tid);
        stage_b(sB2, bT, DD, k0, tid);
#endif
        __syncthreads();
        if (k0 + 32 < DD) {
            load_a_regs(r1, agg, DD, m0, k0 + 32, tid);
            load_a_regs(r2, zin, DD, m0, k0 + 32, tid);
        }

        const bf16x16 a1 = load_a_frag(sA1, wave * 16, lane);
        const bf16x16 a2 = load_a_frag(sA2, wave * 16, lane);
        bf16x16 b1c = load_b_frag(sB1, 0, lane);
        bf16x16 b2c = load_b_frag(sB2, 0, lane);
#pragma unroll
        for (int nt = 0; nt < 16; ++nt) {
            bf16x16 b1n = b1c, b2n = b2c;
            if (nt < 15) {
                b1n = load_b_frag(sB1, (nt + 1) * 16, lane);
                b2n = load_b_frag(sB2, (nt + 1) * 16, lane);
            }
            acc[nt] = __builtin_amdgcn_wmma_f32_16x16x32_bf16(
                false, a1, false, b1c, (short)0, acc[nt], false, false);
            acc[nt] = __builtin_amdgcn_wmma_f32_16x16x32_bf16(
                false, a2, false, b2c, (short)0, acc[nt], false, false);
            b1c = b1n; b2c = b2n;
        }
    }

    const int n  = lane & 15;
    const int mb = (lane >> 4) * 8;
#pragma unroll
    for (int nt = 0; nt < 16; ++nt)
#pragma unroll
        for (int v = 0; v < 8; ++v) {
            const int row = m0 + wave * 16 + mb + v;
            zout[(long)row * DD + nt * 16 + n] = tanhf(acc[nt][v]);
        }
}

// -------------------- host-side orchestration ------------------------------
extern "C" void kernel_launch(void* const* d_in, const int* in_sizes, int n_in,
                              void* d_out, int out_size, void* d_ws, size_t ws_size,
                              hipStream_t stream)
{
    const float* z   = (const float*)d_in[0];   // [NN][DD]
    const float* adj = (const float*)d_in[1];   // [NN][NN]
    const float* deg = (const float*)d_in[2];   // [NN]
    const float* W   = (const float*)d_in[3];   // [2][DD][DD]
    const float* B   = (const float*)d_in[4];   // [2][DD][DD]
    float* out = (float*)d_out;

    char* ws = (char*)d_ws;
    __bf16* zT  = (__bf16*)ws;                                   // 4 MB
    __bf16* wT  = (__bf16*)(ws + (4u << 20));                    // 128 KB
    __bf16* bT  = (__bf16*)(ws + (4u << 20) + (128u << 10));     // 128 KB
    float*  agg = (float*)(ws + (4u << 20) + (256u << 10));      // 8 MB
    float*  z1  = (float*)(ws + (12u << 20) + (256u << 10));     // 8 MB

    const float* zcur = z;
    for (int l = 0; l < 2; ++l) {
        float* znext = (l == 0) ? z1 : out;
        transpose_f32_to_bf16<<<dim3(DD / 32, NN / 32), 256, 0, stream>>>(zcur, zT, NN, DD);
        transpose_f32_to_bf16<<<dim3(DD / 32, DD / 32), 256, 0, stream>>>(W + (long)l * DD * DD, wT, DD, DD);
        transpose_f32_to_bf16<<<dim3(DD / 32, DD / 32), 256, 0, stream>>>(B + (long)l * DD * DD, bT, DD, DD);
        gcn_agg_wmma<<<NN / 64, 128, 0, stream>>>(adj, zT, deg, agg);
        gcn_update_wmma<<<NN / 64, 128, 0, stream>>>(agg, zcur, wT, bT, znext);
        zcur = znext;
    }
}
